// TModel_64536178590337
// MI455X (gfx1250) — matrix operand order repaired and checked by reading, not compiled, not run on hardware
//
#include <hip/hip_runtime.h>
#include <hip/hip_bf16.h>

// ---------------------------------------------------------------------------
// Types for CDNA5 WMMA (wave32, 16x16x32 f16 -> f32)
// ---------------------------------------------------------------------------
typedef __attribute__((ext_vector_type(16))) _Float16 v16h;
typedef __attribute__((ext_vector_type(8)))  float    v8f;
typedef __attribute__((ext_vector_type(4)))  float    f32x4;

#define WMMA_F16(a, b, c) \
  __builtin_amdgcn_wmma_f32_16x16x32_f16(false, (a), false, (b), (short)0, (c), false, false)

// A-fragment gather: A is 16x32 f16 tile at base As (row-major, leading dim lda)
// lanes 0-15 : row m=lane,    elems 0..15 -> K=0..15
// lanes 16-31: row m=lane-16, elems 0..15 -> K=16..31
// Both runs are 8 contiguous halves -> 2x ds_load_b128.
__device__ inline v16h gather_a(const _Float16* As, int lda, int lane) {
  int half = lane >> 4, m = lane & 15;
  v16h a;
#pragma unroll
  for (int e = 0; e < 8; ++e) a[e] = As[m * lda + half * 16 + e];
#pragma unroll
  for (int e = 0; e < 8; ++e) a[8 + e] = As[m * lda + half * 16 + 8 + e];
  return a;
}

__device__ inline float sigm(float z) { return 1.f / (1.f + __expf(-z)); }
__device__ inline float logsig(float z) {
  return fminf(z, 0.f) - log1pf(__expf(-fabsf(z)));
}

// ---------------------------------------------------------------------------
// Generic WMMA GEMM: C[M,N] = A[M,K] @ Bmat + bias
//   Bmat[k][n] = transB ? B[n*ldb + k] : B[k*ldb + n]
// Requirements (all call sites satisfy): M%64==0 (grid-covered), N%64==0,
// K%32==0, lda/ldb multiples of 8, pointers 32B-aligned.
// Block = 256 threads (8 waves), tile 64x64, K-step 32. gridDim.z = batch.
// B is staged in LDS directly in WMMA B-fragment layout:
//   Bs[nt*512 + lane*16 + e] with n = nt*16+(lane&15), k = (lane>>4)*16+e
// so the compute side is one aligned 32-byte v16h LDS load per fragment.
// ---------------------------------------------------------------------------
__global__ __launch_bounds__(256) void gemm_wmma(
    const float* __restrict__ A, const float* __restrict__ B,
    const float* __restrict__ bias, float* __restrict__ C,
    int M, int N, int K, int lda, int ldb, int ldc, int transB,
    long long sA, long long sB, long long sC) {
  A += (size_t)blockIdx.z * sA;
  B += (size_t)blockIdx.z * sB;
  C += (size_t)blockIdx.z * sC;
  __shared__ __attribute__((aligned(32))) _Float16 As[64 * 32];
  __shared__ __attribute__((aligned(32))) _Float16 Bs[4 * 32 * 16];
  int tid = threadIdx.x;
  int bm = blockIdx.y * 64, bn = blockIdx.x * 64;
  int wave = tid >> 5, lane = tid & 31;
  int m0 = (wave & 3) * 16;
  int nt0 = (wave >> 2) * 2;   // two 16-wide N tiles per wave
  v8f acc0 = {}, acc1 = {};
  for (int k0 = 0; k0 < K; k0 += 32) {
    { // stage A tile 64x32 (8 contiguous floats per thread)
      int row = tid >> 2, c0 = (tid & 3) * 8;
      const float* src = A + (size_t)(bm + row) * lda + k0 + c0;
      f32x4 v0 = *(const f32x4*)src;
      f32x4 v1 = *(const f32x4*)(src + 4);
      _Float16* dst = As + row * 32 + c0;
#pragma unroll
      for (int e = 0; e < 4; ++e) dst[e] = (_Float16)v0[e];
#pragma unroll
      for (int e = 0; e < 4; ++e) dst[4 + e] = (_Float16)v1[e];
    }
    if (transB) { // contiguous along k in memory
      int n_l = tid >> 2, k0l = (tid & 3) * 8;
      const float* src = B + (size_t)(bn + n_l) * ldb + k0 + k0l;
      f32x4 v0 = *(const f32x4*)src;
      f32x4 v1 = *(const f32x4*)(src + 4);
      // k run [k0l, k0l+8) stays inside one 16-wide half -> contiguous dest
      int lane_b = ((k0l >> 4) << 4) | (n_l & 15);
      _Float16* dst = Bs + (n_l >> 4) * 512 + lane_b * 16 + (k0l & 15);
#pragma unroll
      for (int e = 0; e < 4; ++e) dst[e] = (_Float16)v0[e];
#pragma unroll
      for (int e = 0; e < 4; ++e) dst[4 + e] = (_Float16)v1[e];
    } else {      // contiguous along n in memory
      int kr = tid >> 3, n0l = (tid & 7) * 8;
      const float* src = B + (size_t)(k0 + kr) * ldb + bn + n0l;
      f32x4 v0 = *(const f32x4*)src;
      f32x4 v1 = *(const f32x4*)(src + 4);
      int khalf = (kr >> 4) << 4, klo = kr & 15;
#pragma unroll
      for (int e = 0; e < 8; ++e) {
        float v = (e < 4) ? v0[e] : v1[e - 4];
        int n = n0l + e;
        Bs[(n >> 4) * 512 + (khalf | (n & 15)) * 16 + klo] = (_Float16)v;
      }
    }
    __syncthreads();
    v16h a  = gather_a(As + m0 * 32, 32, lane);
    v16h b0 = *(const v16h*)(Bs + (nt0 + 0) * 512 + lane * 16);
    v16h b1 = *(const v16h*)(Bs + (nt0 + 1) * 512 + lane * 16);
    acc0 = WMMA_F16(a, b0, acc0);
    acc1 = WMMA_F16(a, b1, acc1);
    __syncthreads();
  }
  // epilogue: C layout per ISA (VGPR r: lanes0-15 M=r, lanes16-31 M=r+8)
  int mrow  = bm + m0 + ((lane >> 4) << 3);
  int ncol0 = bn + nt0 * 16 + (lane & 15);
  int ncol1 = ncol0 + 16;
  float bias0 = bias ? bias[ncol0] : 0.f;
  float bias1 = bias ? bias[ncol1] : 0.f;
#pragma unroll
  for (int r = 0; r < 8; ++r) {
    int m = mrow + r;
    if (m < M) {
      C[(size_t)m * ldc + ncol0] = acc0[r] + bias0;
      C[(size_t)m * ldc + ncol1] = acc1[r] + bias1;
    }
  }
}

// ---------------------------------------------------------------------------
// Pack Whh [1024(n) x 256(k)] f32 into WMMA B-fragment layout:
// dst[((ntile*8 + kstep)*32 + lane)*16 + e] with n = ntile*16 + (lane&15),
// k = kstep*32 + (lane>>4)*16 + e  -> each lane loads one contiguous v16h.
// ---------------------------------------------------------------------------
__global__ void pack_whh(const float* __restrict__ src, _Float16* __restrict__ dst) {
  int idx = blockIdx.x * blockDim.x + threadIdx.x;
  if (idx >= 64 * 8 * 32 * 16) return;
  int e = idx & 15;
  int lane = (idx >> 4) & 31;
  int ks = (idx >> 9) & 7;
  int nt = idx >> 12;
  int n = nt * 16 + (lane & 15);
  int k = ks * 32 + (lane >> 4) * 16 + e;
  dst[idx] = (_Float16)src[n * 256 + k];
}

// ---------------------------------------------------------------------------
// Persistent LSTM scan: one direction per block (512 thr = 16 waves).
// Per step: g = xg[:,t,:] + h @ Whh^T via 16x16x32 WMMA (64 N-tiles, K=256),
// then elementwise gates; h/c live in LDS, Whh fragments stream from L2.
// ---------------------------------------------------------------------------
struct ScanDir {
  const float* xg;        // [4096, 1024]
  const _Float16* whhp;   // packed fragments
  float* out;             // hidden output base
  int ldo;                // leading dim of out
  int col_off;            // column offset (0 fwd / 256 bwd in concat)
  int rev;                // scan direction
};
struct ScanArgs { ScanDir d[4]; };

__global__ __launch_bounds__(512) void lstm_scan(ScanArgs args) {
  ScanDir dd = args.d[blockIdx.x];
  __shared__ __attribute__((aligned(32))) _Float16 h_lds[16 * 256]; // rows 8-15 zero
  __shared__ float c_lds[8 * 256];
  __shared__ float g_lds[8 * 1024];
  int tid = threadIdx.x;
  for (int i = tid; i < 16 * 256; i += 512) h_lds[i] = (_Float16)0.f;
  for (int i = tid; i < 8 * 256; i += 512) c_lds[i] = 0.f;
  __syncthreads();
  int wave = tid >> 5, lane = tid & 31;
  for (int s = 0; s < 512; ++s) {
    int t = dd.rev ? (511 - s) : s;
    v8f acc[4] = {};
#pragma unroll
    for (int ks = 0; ks < 8; ++ks) {
      v16h a = gather_a(h_lds + ks * 32, 256, lane);
#pragma unroll
      for (int j = 0; j < 4; ++j) {
        int ntile = wave * 4 + j;
        v16h b = *(const v16h*)(dd.whhp + ((size_t)((ntile * 8 + ks) * 32 + lane)) * 16);
        acc[j] = WMMA_F16(a, b, acc[j]);
      }
    }
    if (lane < 16) { // rows M=0..7 (valid batch) live in lanes 0-15
#pragma unroll
      for (int j = 0; j < 4; ++j) {
        int n = (wave * 4 + j) * 16 + lane;
#pragma unroll
        for (int r = 0; r < 8; ++r) g_lds[r * 1024 + n] = acc[j][r];
      }
    }
    __syncthreads();
    // gates: 8 batches x 256 hidden = 2048 elems
    for (int e = tid; e < 2048; e += 512) {
      int b = e >> 8, n = e & 255;
      const float* xgp = dd.xg + ((size_t)(b * 512 + t)) * 1024;
      float gi = xgp[n]       + g_lds[b * 1024 + n];
      float gf = xgp[256 + n] + g_lds[b * 1024 + 256 + n];
      float gg = xgp[512 + n] + g_lds[b * 1024 + 512 + n];
      float go = xgp[768 + n] + g_lds[b * 1024 + 768 + n];
      float c = sigm(gf) * c_lds[b * 256 + n] + sigm(gi) * tanhf(gg);
      float h = sigm(go) * tanhf(c);
      c_lds[b * 256 + n] = c;
      h_lds[b * 256 + n] = (_Float16)h;
      dd.out[((size_t)(b * 512 + t)) * dd.ldo + dd.col_off + n] = h;
    }
    __syncthreads();
  }
}

// ---------------------------------------------------------------------------
// Small helper kernels
// ---------------------------------------------------------------------------
// out[r] = sum_k src[r*ld+k] * (w ? w[k] : 1) + (biasp ? biasp[0] : 0)
__global__ void dot_rows(const float* __restrict__ src, const float* __restrict__ w,
                         const float* __restrict__ biasp, float* __restrict__ out,
                         int rows, int K, int ld) {
  int r = blockIdx.x * blockDim.x + threadIdx.x;
  if (r >= rows) return;
  const float* p = src + (size_t)r * ld;
  float s = 0.f;
  for (int k = 0; k < K; ++k) s += p[k] * (w ? w[k] : 1.f);
  out[r] = s + (biasp ? biasp[0] : 0.f);
}

// xgd[row,g] = (ch[l-1]+ch[l]+ch[l+1]) * wsum[g] + bd_b[g]  (tri is rank-1!)
__global__ void win_xgdec(const float* __restrict__ ch, const float* __restrict__ wsum,
                          const float* __restrict__ bdb, float* __restrict__ xgd) {
  int idx = blockIdx.x * blockDim.x + threadIdx.x;
  if (idx >= 4096 * 1024) return;
  int row = idx >> 10, g = idx & 1023;
  int l = row & 511;
  float w = ch[row];
  if (l > 0)   w += ch[row - 1];
  if (l < 511) w += ch[row + 1];
  xgd[idx] = w * wsum[g] + bdb[g];
}

// softmax over i of (scores[b,j,i] + bv[b,i]); apply triu mask (i>=j); rowsum
__global__ __launch_bounds__(256) void softmax_mask(
    const float* __restrict__ bv, float* __restrict__ S, float* __restrict__ rowsum) {
  int b = blockIdx.x >> 9, j = blockIdx.x & 511;
  float* row = S + ((size_t)b * 512 + j) * 512;
  const float* bvb = bv + b * 512;
  __shared__ float red[256];
  int tid = threadIdx.x;
  float v0 = row[tid] + bvb[tid];
  float v1 = row[tid + 256] + bvb[tid + 256];
  red[tid] = fmaxf(v0, v1); __syncthreads();
  for (int s = 128; s > 0; s >>= 1) { if (tid < s) red[tid] = fmaxf(red[tid], red[tid + s]); __syncthreads(); }
  float mx = red[0]; __syncthreads();
  float e0 = __expf(v0 - mx), e1 = __expf(v1 - mx);
  red[tid] = e0 + e1; __syncthreads();
  for (int s = 128; s > 0; s >>= 1) { if (tid < s) red[tid] += red[tid + s]; __syncthreads(); }
  float inv = 1.f / red[0]; __syncthreads();
  float w0 = (tid >= j)       ? e0 * inv : 0.f;
  float w1 = (tid + 256 >= j) ? e1 * inv : 0.f;
  row[tid] = w0; row[tid + 256] = w1;
  red[tid] = w0 + w1; __syncthreads();
  for (int s = 128; s > 0; s >>= 1) { if (tid < s) red[tid] += red[tid + s]; __syncthreads(); }
  if (tid == 0) rowsum[b * 512 + j] = red[0];
}

// out = log_sigmoid(a + b2) * h
__global__ void ew_gate(const float* __restrict__ a, const float* __restrict__ b2,
                        const float* __restrict__ h, float* __restrict__ out, int n) {
  int i = blockIdx.x * blockDim.x + threadIdx.x;
  if (i < n) out[i] = logsig(a[i] + b2[i]) * h[i];
}

// ner head over concat [x | bres | tres | seg1 | rowsum*bh] @ nfw^T + nfb
__global__ void ner_head(const float* __restrict__ x, const float* __restrict__ bres,
                         const float* __restrict__ tres, const float* __restrict__ seg1,
                         const float* __restrict__ bh, const float* __restrict__ rowsum,
                         const float* __restrict__ nfw, const float* __restrict__ nfb,
                         float* __restrict__ out) {
  int idx = blockIdx.x * blockDim.x + threadIdx.x;
  if (idx >= 4096 * 9) return;
  int row = idx / 9, o = idx % 9;
  const float* w = nfw + (size_t)o * 2816;
  float s = nfb[o];
  const float* xp = x + (size_t)row * 768;
  for (int k = 0; k < 768; ++k) s += xp[k] * w[k];
  const float* bp = bres + (size_t)row * 512;
  for (int k = 0; k < 512; ++k) s += bp[k] * w[768 + k];
  const float* tp = tres + (size_t)row * 512;
  for (int k = 0; k < 512; ++k) s += tp[k] * w[1280 + k];
  const float* s1 = seg1 + (size_t)row * 512;
  for (int k = 0; k < 512; ++k) s += s1[k] * w[1792 + k];
  float rs = rowsum[row];
  const float* hp = bh + (size_t)row * 512;
  for (int k = 0; k < 512; ++k) s += rs * hp[k] * w[2304 + k];
  out[idx] = s;
}

// tiny heads: out[row,o] = dot(src[row], w[o]) + b[o]
__global__ void small_head(const float* __restrict__ src, const float* __restrict__ w,
                           const float* __restrict__ b, float* __restrict__ out,
                           int rows, int K, int no) {
  int idx = blockIdx.x * blockDim.x + threadIdx.x;
  if (idx >= rows * no) return;
  int row = idx / no, o = idx % no;
  const float* p = src + (size_t)row * K;
  const float* wp = w + (size_t)o * K;
  float s = b[o];
  for (int k = 0; k < K; ++k) s += p[k] * wp[k];
  out[idx] = s;
}

// ---------------------------------------------------------------------------
// Orchestration
// ---------------------------------------------------------------------------
extern "C" void kernel_launch(void* const* d_in, const int* in_sizes, int n_in,
                              void* d_out, int out_size, void* d_ws, size_t ws_size,
                              hipStream_t stream) {
  (void)in_sizes; (void)n_in; (void)out_size; (void)ws_size;
  const float* x      = (const float*)d_in[0];
  const float* be_Wih = (const float*)d_in[1];
  const float* be_Whh = (const float*)d_in[2];
  const float* be_b   = (const float*)d_in[3];
  const float* tl_Wih = (const float*)d_in[4];
  const float* tl_Whh = (const float*)d_in[5];
  const float* tl_b   = (const float*)d_in[6];
  const float* bd_Wih = (const float*)d_in[7];
  const float* bd_Whh = (const float*)d_in[8];
  const float* bd_b   = (const float*)d_in[9];
  const float* Wb     = (const float*)d_in[10];
  // d_in[11]=bb, d_in[12]=Uw, d_in[13]=Ub: constant along softmax axis -> cancel
  const float* Vw     = (const float*)d_in[14];
  const float* Vb     = (const float*)d_in[15];
  const float* bf0w   = (const float*)d_in[16];
  const float* bf0b   = (const float*)d_in[17];
  const float* bf1w   = (const float*)d_in[18];
  const float* bf1b   = (const float*)d_in[19];
  const float* bfcw   = (const float*)d_in[20];
  const float* bfcb   = (const float*)d_in[21];
  const float* tf0w   = (const float*)d_in[22];
  const float* tf0b   = (const float*)d_in[23];
  const float* tf1w   = (const float*)d_in[24];
  const float* tf1b   = (const float*)d_in[25];
  const float* tfcw   = (const float*)d_in[26];
  const float* tfcb   = (const float*)d_in[27];
  const float* nfw    = (const float*)d_in[28];
  const float* nfb    = (const float*)d_in[29];
  float* out = (float*)d_out;
  float* ws  = (float*)d_ws;

  // ---- workspace layout (floats) ----
  const size_t XG   = 0;                 // 5 x 4096x1024 (xg per dir; reused after scans)
  const size_t BH   = 20971520;          // 4096 x 512
  const size_t TH   = 23068672;          // 4096 x 512
  const size_t DEC  = 25165824;          // 4096 x 256
  const size_t WHHP = 26214400;          // 5 x 262144 halfs (as 655360 floats)
  const size_t CH   = 26869760;          // 4096
  const size_t WSUM = 26873856;          // 1024
  const size_t BV   = 26874880;          // 4096
  const size_t RSUM = 26878976;          // 4096
  // reuse of XG region after all scans:
  const size_t DECW = 0;                 // 4096 x 512
  const size_t SCR  = 2097152;           // 8 x 512 x 512 (scores -> wadj in place)
  const size_t SEG1 = 4194304;           // 4096 x 512
  const size_t BF0  = 6291456;
  const size_t BF1  = 8388608;
  const size_t TF0  = 10485760;
  const size_t TF1  = 12582912;
  const size_t BRES = 14680064;
  const size_t TRES = 16777216;

  _Float16* whhp = (_Float16*)(ws + WHHP);

  auto gemm = [&](const float* A, const float* B, const float* bias, float* C,
                  int M, int N, int K, int lda, int ldb, int ldc, int transB,
                  long long sA, long long sB, long long sC, int batch) {
    dim3 g((N + 63) / 64, (M + 63) / 64, batch);
    gemm_wmma<<<g, 256, 0, stream>>>(A, B, bias, C, M, N, K, lda, ldb, ldc,
                                     transB, sA, sB, sC);
  };

  // 1) input projections xg = x @ Wih^T + b  (M=4096, N=1024, K=768)
  gemm(x, be_Wih,            be_b,        ws + XG + 0 * 4194304, 4096, 1024, 768, 768, 768, 1024, 1, 0, 0, 0, 1);
  gemm(x, be_Wih + 1024*768, be_b + 1024, ws + XG + 1 * 4194304, 4096, 1024, 768, 768, 768, 1024, 1, 0, 0, 0, 1);
  gemm(x, tl_Wih,            tl_b,        ws + XG + 2 * 4194304, 4096, 1024, 768, 768, 768, 1024, 1, 0, 0, 0, 1);
  gemm(x, tl_Wih + 1024*768, tl_b + 1024, ws + XG + 3 * 4194304, 4096, 1024, 768, 768, 768, 1024, 1, 0, 0, 0, 1);

  // 2) pack Whh into WMMA fragment layout (5 directions)
  const float* whh_src[5] = { be_Whh, be_Whh + 1024*256, tl_Whh, tl_Whh + 1024*256, bd_Whh };
  for (int d = 0; d < 5; ++d)
    pack_whh<<<(64*8*32*16 + 255) / 256, 256, 0, stream>>>(whh_src[d], whhp + (size_t)d * 262144);

  // 3) four encoder directions, one block each (concurrent)
  {
    ScanArgs sa;
    sa.d[0] = { ws + XG + 0 * 4194304, whhp + 0 * 262144, ws + BH, 512,   0, 0 };
    sa.d[1] = { ws + XG + 1 * 4194304, whhp + 1 * 262144, ws + BH, 512, 256, 1 };
    sa.d[2] = { ws + XG + 2 * 4194304, whhp + 2 * 262144, ws + TH, 512,   0, 0 };
    sa.d[3] = { ws + XG + 3 * 4194304, whhp + 3 * 262144, ws + TH, 512, 256, 1 };
    lstm_scan<<<4, 512, 0, stream>>>(sa);
  }

  // 4) ch = bh.sum(-1); wsum = rowsum(bd_Wih); bv = bh@Vw^T + Vb
  dot_rows<<<(4096 + 255) / 256, 256, 0, stream>>>(ws + BH, nullptr, nullptr, ws + CH, 4096, 512, 512);
  dot_rows<<<(1024 + 255) / 256, 256, 0, stream>>>(bd_Wih, nullptr, nullptr, ws + WSUM, 1024, 512, 512);
  dot_rows<<<(4096 + 255) / 256, 256, 0, stream>>>(ws + BH, Vw, Vb, ws + BV, 4096, 512, 512);

  // 5) decoder xg (tri is rank-1 -> no GEMM)
  win_xgdec<<<(4096 * 1024 + 255) / 256, 256, 0, stream>>>(ws + CH, ws + WSUM, bd_b, ws + XG + 4 * 4194304);

  // 6) decoder scan
  {
    ScanArgs sa;
    ScanDir d0 = { ws + XG + 4 * 4194304, whhp + 4 * 262144, ws + DEC, 256, 0, 0 };
    sa.d[0] = d0; sa.d[1] = d0; sa.d[2] = d0; sa.d[3] = d0;
    lstm_scan<<<1, 512, 0, stream>>>(sa);
  }

  // 7) decW = dec @ Wb[:,:,0]  (Wb is [K=256][N=512], transB=0)
  gemm(ws + DEC, Wb, nullptr, ws + DECW, 4096, 512, 256, 256, 512, 512, 0, 0, 0, 0, 1);

  // 8) scores[b] = decW[b] @ bh[b]^T  (batched 512x512x512)
  gemm(ws + DECW, ws + BH, nullptr, ws + SCR, 512, 512, 512, 512, 512, 512, 1,
       512 * 512, 512 * 512, 512 * 512, 8);

  // 9) softmax over i (+bv), triu mask, rowsum -> wadj in place
  softmax_mask<<<4096, 256, 0, stream>>>(ws + BV, ws + SCR, ws + RSUM);

  // 10) seg1[b] = wadj[b] @ bh[b]
  gemm(ws + SCR, ws + BH, nullptr, ws + SEG1, 512, 512, 512, 512, 512, 512, 0,
       512 * 512, 512 * 512, 512 * 512, 8);

  // 11) boundary / type gates
  gemm(x,       bf0w, bf0b, ws + BF0, 4096, 512, 768, 768, 768, 512, 1, 0, 0, 0, 1);
  gemm(ws + BH, bf1w, bf1b, ws + BF1, 4096, 512, 512, 512, 512, 512, 1, 0, 0, 0, 1);
  gemm(x,       tf0w, tf0b, ws + TF0, 4096, 512, 768, 768, 768, 512, 1, 0, 0, 0, 1);
  gemm(ws + TH, tf1w, tf1b, ws + TF1, 4096, 512, 512, 512, 512, 512, 1, 0, 0, 0, 1);
  ew_gate<<<(2097152 + 255) / 256, 256, 0, stream>>>(ws + BF0, ws + BF1, ws + BH, ws + BRES, 2097152);
  ew_gate<<<(2097152 + 255) / 256, 256, 0, stream>>>(ws + TF0, ws + TF1, ws + TH, ws + TRES, 2097152);

  // 12) output heads: ner [4096,9] | bfc [4096,2] | tfc [4096,7]
  ner_head<<<(4096 * 9 + 255) / 256, 256, 0, stream>>>(
      x, ws + BRES, ws + TRES, ws + SEG1, ws + BH, ws + RSUM, nfw, nfb, out);
  small_head<<<(4096 * 2 + 255) / 256, 256, 0, stream>>>(ws + BH, bfcw, bfcb, out + 36864, 4096, 512, 2);
  small_head<<<(4096 * 7 + 255) / 256, 256, 0, stream>>>(ws + TH, tfcw, tfcb, out + 45056, 4096, 512, 7);
}
